// full_graphConv_48215302865680
// MI455X (gfx1250) — matrix operand order repaired and checked by reading, not compiled, not run on hardware
//
#include <hip/hip_runtime.h>

typedef __bf16 bf16;
typedef __attribute__((ext_vector_type(16))) __bf16 v16bf;
typedef __attribute__((ext_vector_type(8)))  __bf16 v8bf;
typedef __attribute__((ext_vector_type(8)))  float  v8f;

#define EPS 1e-5f

__device__ __forceinline__ bf16 f2bf(float f) {
    unsigned u = __float_as_uint(f);
    unsigned r = (u + 0x7FFFu + ((u >> 16) & 1u)) >> 16;   // round-to-nearest-even
    unsigned short s = (unsigned short)r;
    return __builtin_bit_cast(bf16, s);
}
__device__ __forceinline__ float bf2f(bf16 b) {
    unsigned short s = __builtin_bit_cast(unsigned short, b);
    return __uint_as_float(((unsigned)s) << 16);
}

// ---------------- weight prep: transpose + fp32 -> bf16 (Wt[n*K+k] = W[k*N+n]) ----
__global__ void k_wt_prep(const float* __restrict__ W, bf16* __restrict__ Wt, int K, int Nc) {
    int idx = blockIdx.x * blockDim.x + threadIdx.x;
    if (idx >= K * Nc) return;
    int n = idx / K;
    int k = idx - n * K;
    Wt[idx] = f2bf(W[(size_t)k * Nc + n]);
}

__global__ void k_fill(float* __restrict__ p, float v, int n) {
    int i = blockIdx.x * blockDim.x + threadIdx.x;
    if (i < n) p[i] = v;
}

__global__ void k_f2bf(const float* __restrict__ in, bf16* __restrict__ out, int n) {
    int i = blockIdx.x * blockDim.x + threadIdx.x;
    if (i < n) out[i] = f2bf(in[i]);
}

// ---------------- degree: deg[d] += ew[e]  (deg pre-filled with 1.0 self-loop) ----
__global__ void k_edge_deg(const long long* __restrict__ dst, const float* __restrict__ ew,
                           float* __restrict__ deg, int E) {
    int e = blockIdx.x * blockDim.x + threadIdx.x;
    if (e >= E) return;
    atomicAdd(&deg[(int)dst[e]], ew[e]);
}

// ---------------- BN stats: per-channel sum / sumsq of (relu?) X --------------------
__global__ void k_bn_stats(const float* __restrict__ X, float* __restrict__ stats,
                           int Nn, int C, int relu) {
    int c = threadIdx.x;                 // blockDim.x == C
    float s = 0.f, q = 0.f;
    for (int r = blockIdx.x; r < Nn; r += gridDim.x) {
        float v = X[(size_t)r * C + c];
        if (relu) v = fmaxf(v, 0.f);
        s += v; q += v * v;
    }
    atomicAdd(&stats[c], s);
    atomicAdd(&stats[C + c], q);
}

// ---------------- BN apply: y = g*(relu?(x)-m)*rsqrt(v+eps)+b -> bf16 and/or f32 ----
__global__ void k_bn_apply(const float* __restrict__ X, const float* __restrict__ stats,
                           const float* __restrict__ g, const float* __restrict__ be,
                           bf16* __restrict__ obf, float* __restrict__ of,
                           int Nn, int C, int relu) {
    int idx = blockIdx.x * blockDim.x + threadIdx.x;
    if (idx >= Nn * C) return;
    int c = idx % C;
    float invN = 1.0f / (float)Nn;
    float m = stats[c] * invN;
    float var = stats[C + c] * invN - m * m;
    float v = X[idx];
    if (relu) v = fmaxf(v, 0.f);
    float y = g[c] * (v - m) * rsqrtf(var + EPS) + be[c];
    if (obf) obf[idx] = f2bf(y);
    if (of)  of[idx]  = y;
}

// ---------------- GCN self-loop term: Agg = XW/deg + b ------------------------------
__global__ void k_gcn_self(const float* __restrict__ XW, const float* __restrict__ deg,
                           const float* __restrict__ b, float* __restrict__ Agg,
                           int total, int H) {
    int idx = blockIdx.x * blockDim.x + threadIdx.x;
    if (idx >= total) return;
    int row = idx / H;
    int c = idx - row * H;
    Agg[idx] = XW[idx] / deg[row] + b[c];
}

// ---------------- GCN edge scatter: Agg[dst] += dinv[s]*ew*dinv[d] * XW[src] --------
__global__ void k_gcn_scatter(const long long* __restrict__ src, const long long* __restrict__ dst,
                              const float* __restrict__ ew, const float* __restrict__ deg,
                              const float* __restrict__ XW, float* __restrict__ Agg,
                              int E, int H) {
    int idx = blockIdx.x * blockDim.x + threadIdx.x;
    int per = H >> 2;                       // 4 channels per thread (float4)
    if (idx >= E * per) return;
    int e = idx / per;
    int c = (idx - e * per) << 2;
    int s = (int)src[e], d = (int)dst[e];
    float nrm = rsqrtf(deg[s]) * ew[e] * rsqrtf(deg[d]);
    const float4 v = *(const float4*)(XW + (size_t)s * H + c);
    float* p = Agg + (size_t)d * H + c;
    atomicAdd(p + 0, nrm * v.x);
    atomicAdd(p + 1, nrm * v.y);
    atomicAdd(p + 2, nrm * v.z);
    atomicAdd(p + 3, nrm * v.w);
}

// ---------------- GraphConv edge scatter: Agg[dst] += ew * H[src]  (bf16 gather) ----
__global__ void k_graph_scatter(const long long* __restrict__ src, const long long* __restrict__ dst,
                                const float* __restrict__ ew, const bf16* __restrict__ Hb,
                                float* __restrict__ Agg, int E, int H) {
    int idx = blockIdx.x * blockDim.x + threadIdx.x;
    int per = H >> 3;                       // 8 channels per thread (16B bf16 load)
    if (idx >= E * per) return;
    int e = idx / per;
    int c = (idx - e * per) << 3;
    int s = (int)src[e], d = (int)dst[e];
    float w = ew[e];
    const v8bf hv = *(const v8bf*)(Hb + (size_t)s * H + c);
    float* p = Agg + (size_t)d * H + c;
#pragma unroll
    for (int i = 0; i < 8; i++) atomicAdd(p + i, w * bf2f(hv[i]));
}

// ---------------- bf16 WMMA GEMM: C[M,N] (=|+=) A[M,K] @ Bt[N,K]^T (+bias) ----------
// Block = 8 waves = 8 M-tiles x one 64-col N-strip. The B strip (64 x K bf16) is
// staged once into LDS (padded stride K+8 -> ds_load_b128 hits all 64 banks
// conflict-free) and shared by all 8 waves. The A fragment is software-pipelined:
// the next k-step's global loads are issued before the current WMMA block, so the
// global latency is hidden behind 4 WMMAs + the DS fragment fetches.
#define BKP_MAX 264                         // max K (256) + 8 pad elements
__global__ __launch_bounds__(256)
void k_gemm_bf16(const bf16* __restrict__ A, const bf16* __restrict__ Bt,
                 const float* __restrict__ bias, float* __restrict__ C,
                 int M, int N, int K, int accumulate) {
    __shared__ __align__(16) bf16 Bs[64 * BKP_MAX];   // 33792 B of the 320KB/WGP LDS
    const int lane = threadIdx.x & 31;
    const int wave = threadIdx.x >> 5;
    const int nStrips = N >> 6;
    const int strip   = blockIdx.x % nStrips;
    const int mchunk  = blockIdx.x / nStrips;
    const int tN = strip << 6;
    const int tM = (mchunk * 8 + wave) << 4;
    const int KP = K + 8;

    // ---- cooperative stage of B strip into LDS (all 256 threads) ----
    const int chunksPerCol = K >> 3;              // 8 bf16 (16B) per chunk
    const int totalChunks = 64 * chunksPerCol;
    for (int idx = threadIdx.x; idx < totalChunks; idx += 256) {
        int c  = idx / chunksPerCol;
        int kc = (idx - c * chunksPerCol) << 3;
        *(v8bf*)(&Bs[c * KP + kc]) = *(const v8bf*)(Bt + (size_t)(tN + c) * K + kc);
    }
    __syncthreads();

    if (tM < M) {                                  // wave-uniform guard: EXEC all-1s inside
        const int l15 = lane & 15;
        const int hi  = lane >> 4;
        const int aoff = hi << 3;                  // A frag: K chunks {aoff..+7},{aoff+16..+23}
        const int boff = hi << 3;                  // B frag: K chunks {boff..+7},{boff+16..+23}
        const bf16* pa = A + (long)(tM + l15) * K + aoff;

        v8f acc[4] = {};
        v8bf alo = *(const v8bf*)(pa);             // prologue: A(k=0) in flight
        v8bf ahi = *(const v8bf*)(pa + 16);
        for (int k0 = 0; k0 < K; k0 += 32) {
            v16bf a;
#pragma unroll
            for (int i = 0; i < 8; i++) { a[i] = alo[i]; a[i + 8] = ahi[i]; }
            if (k0 + 32 < K) {                     // wave-uniform: prefetch next A fragment
                const bf16* pn = pa + k0 + 32;
                alo = *(const v8bf*)(pn);
                ahi = *(const v8bf*)(pn + 16);
            }

            v16bf bfrag[4];
#pragma unroll
            for (int j = 0; j < 4; j++) {          // all B fragments first (one DS wait)
                const bf16* pb = &Bs[(j * 16 + l15) * KP + k0 + boff];
                v8bf blo = *(const v8bf*)(pb);
                v8bf bhi = *(const v8bf*)(pb + 16);
#pragma unroll
                for (int i = 0; i < 8; i++) { bfrag[j][i] = blo[i]; bfrag[j][i + 8] = bhi[i]; }
            }
#pragma unroll
            for (int j = 0; j < 4; j++) {          // then 4 back-to-back WMMAs
                acc[j] = __builtin_amdgcn_wmma_f32_16x16x32_bf16(
                    false, a, false, bfrag[j], (short)0, acc[j], false, false);
            }
        }

        const int mBase = tM + (hi << 3);          // C/D: lanes 0-15 -> M=r, 16-31 -> M=8+r
#pragma unroll
        for (int r = 0; r < 8; r++) {
            float* pc = C + (long)(mBase + r) * N + tN + l15;
#pragma unroll
            for (int j = 0; j < 4; j++) {
                float v = acc[j][r];
                if (bias) v += bias[tN + j * 16 + l15];
                if (accumulate) pc[j * 16] += v;
                else            pc[j * 16]  = v;
            }
        }
    }
}

// ======================================================================================
extern "C" void kernel_launch(void* const* d_in, const int* in_sizes, int n_in,
                              void* d_out, int out_size, void* d_ws, size_t ws_size,
                              hipStream_t stream) {
    const float*     x      = (const float*)d_in[0];
    const long long* ei     = (const long long*)d_in[1];
    const float*     ew     = (const float*)d_in[2];
    const float*     W1     = (const float*)d_in[3];
    const float*     b1     = (const float*)d_in[4];
    const float*     Wrel2  = (const float*)d_in[5];
    const float*     brel2  = (const float*)d_in[6];
    const float*     Wroot2 = (const float*)d_in[7];
    const float*     Wrel3  = (const float*)d_in[8];
    const float*     brel3  = (const float*)d_in[9];
    const float*     Wroot3 = (const float*)d_in[10];
    const float* g0 = (const float*)d_in[11], *be0 = (const float*)d_in[12];
    const float* g1 = (const float*)d_in[13], *be1 = (const float*)d_in[14];
    const float* g2 = (const float*)d_in[15], *be2 = (const float*)d_in[16];
    const float* g3 = (const float*)d_in[17], *be3 = (const float*)d_in[18];

    const int CIN = 128, H1 = 256, H2 = 128;
    const int Nn = in_sizes[0] / CIN;       // 50000
    const int E  = in_sizes[2];             // 800000
    const long long* srcI = ei;
    const long long* dstI = ei + E;

    // ---- workspace carve-up (256B aligned) ----
    char* ws = (char*)d_ws;
    size_t off = 0;
    auto alloc = [&](size_t bytes) -> char* {
        char* p = ws + off;
        off = (off + bytes + 255) & ~(size_t)255;
        return p;
    };
    bf16*  Wt1     = (bf16*)alloc((size_t)CIN * H1 * 2);
    bf16*  Wrel2t  = (bf16*)alloc((size_t)H1 * H1 * 2);
    bf16*  Wroot2t = (bf16*)alloc((size_t)H1 * H1 * 2);
    bf16*  Wrel3t  = (bf16*)alloc((size_t)H1 * H2 * 2);
    bf16*  Wroot3t = (bf16*)alloc((size_t)H1 * H2 * 2);
    float* deg     = (float*)alloc((size_t)Nn * 4);
    float* stats   = (float*)alloc(2 * 256 * 4);
    bf16*  Hbf     = (bf16*)alloc((size_t)Nn * H1 * 2);   // current layer features (bf16)
    bf16*  Gbf     = (bf16*)alloc((size_t)Nn * H1 * 2);   // aggregated features (bf16)
    float* AggF    = (float*)alloc((size_t)Nn * H1 * 4);  // fp32 scatter accumulator
    float* OutF    = (float*)alloc((size_t)Nn * H1 * 4);  // fp32 pre-BN layer output

    auto cdiv = [](int a, int b) { return (a + b - 1) / b; };
    // block = 8 M-tiles x one 64-col strip
    auto gemmBlocks = [&](int M, int N) { return cdiv(M >> 4, 8) * (N >> 6); };

    // ---- weight prep (bf16, pre-transposed) ----
    k_wt_prep<<<cdiv(CIN * H1, 256), 256, 0, stream>>>(W1,     Wt1,     CIN, H1);
    k_wt_prep<<<cdiv(H1 * H1, 256),  256, 0, stream>>>(Wrel2,  Wrel2t,  H1,  H1);
    k_wt_prep<<<cdiv(H1 * H1, 256),  256, 0, stream>>>(Wroot2, Wroot2t, H1,  H1);
    k_wt_prep<<<cdiv(H1 * H2, 256),  256, 0, stream>>>(Wrel3,  Wrel3t,  H1,  H2);
    k_wt_prep<<<cdiv(H1 * H2, 256),  256, 0, stream>>>(Wroot3, Wroot3t, H1,  H2);

    // ---- degrees: deg = segsum(ew, dst) + 1 ----
    k_fill<<<cdiv(Nn, 256), 256, 0, stream>>>(deg, 1.0f, Nn);
    k_edge_deg<<<cdiv(E, 256), 256, 0, stream>>>(dstI, ew, deg, E);

    // ---- BN0 on x -> Hbf (bf16) ----
    hipMemsetAsync(stats, 0, 2 * 256 * 4, stream);
    k_bn_stats<<<512, CIN, 0, stream>>>(x, stats, Nn, CIN, 0);
    k_bn_apply<<<cdiv(Nn * CIN, 256), 256, 0, stream>>>(x, stats, g0, be0,
                                                        Hbf, nullptr, Nn, CIN, 0);

    // ---- layer 1: GCNConv(128->256) ----
    k_gemm_bf16<<<gemmBlocks(Nn, H1), 256, 0, stream>>>(Hbf, Wt1, nullptr, OutF,
                                                        Nn, H1, CIN, 0);   // XW
    k_gcn_self<<<cdiv(Nn * H1, 256), 256, 0, stream>>>(OutF, deg, b1, AggF, Nn * H1, H1);
    k_gcn_scatter<<<cdiv(E * (H1 >> 2), 256), 256, 0, stream>>>(srcI, dstI, ew, deg,
                                                                OutF, AggF, E, H1);
    hipMemsetAsync(stats, 0, 2 * 256 * 4, stream);
    k_bn_stats<<<512, H1, 0, stream>>>(AggF, stats, Nn, H1, 1);
    k_bn_apply<<<cdiv(Nn * H1, 256), 256, 0, stream>>>(AggF, stats, g1, be1,
                                                       Hbf, nullptr, Nn, H1, 1);

    // ---- layer 2: GraphConv(256->256) ----
    hipMemsetAsync(AggF, 0, (size_t)Nn * H1 * 4, stream);
    k_graph_scatter<<<cdiv(E * (H1 >> 3), 256), 256, 0, stream>>>(srcI, dstI, ew, Hbf,
                                                                  AggF, E, H1);
    k_f2bf<<<cdiv(Nn * H1, 256), 256, 0, stream>>>(AggF, Gbf, Nn * H1);
    k_gemm_bf16<<<gemmBlocks(Nn, H1), 256, 0, stream>>>(Gbf, Wrel2t, brel2, OutF,
                                                        Nn, H1, H1, 0);
    k_gemm_bf16<<<gemmBlocks(Nn, H1), 256, 0, stream>>>(Hbf, Wroot2t, nullptr, OutF,
                                                        Nn, H1, H1, 1);
    hipMemsetAsync(stats, 0, 2 * 256 * 4, stream);
    k_bn_stats<<<512, H1, 0, stream>>>(OutF, stats, Nn, H1, 1);
    k_bn_apply<<<cdiv(Nn * H1, 256), 256, 0, stream>>>(OutF, stats, g2, be2,
                                                       Hbf, nullptr, Nn, H1, 1);

    // ---- layer 3: GraphConv(256->128) -> d_out ----
    hipMemsetAsync(AggF, 0, (size_t)Nn * H1 * 4, stream);
    k_graph_scatter<<<cdiv(E * (H1 >> 3), 256), 256, 0, stream>>>(srcI, dstI, ew, Hbf,
                                                                  AggF, E, H1);
    k_f2bf<<<cdiv(Nn * H1, 256), 256, 0, stream>>>(AggF, Gbf, Nn * H1);
    k_gemm_bf16<<<gemmBlocks(Nn, H2), 256, 0, stream>>>(Gbf, Wrel3t, brel3, OutF,
                                                        Nn, H2, H1, 0);
    k_gemm_bf16<<<gemmBlocks(Nn, H2), 256, 0, stream>>>(Hbf, Wroot3t, nullptr, OutF,
                                                        Nn, H2, H1, 1);
    hipMemsetAsync(stats, 0, 2 * 256 * 4, stream);
    k_bn_stats<<<512, H2, 0, stream>>>(OutF, stats, Nn, H2, 1);
    k_bn_apply<<<cdiv(Nn * H2, 256), 256, 0, stream>>>(OutF, stats, g3, be3,
                                                       nullptr, (float*)d_out, Nn, H2, 1);
}
